// LSTM_67233418051872
// MI455X (gfx1250) — matrix operand order repaired and checked by reading, not compiled, not run on hardware
//
#include <hip/hip_runtime.h>
#include <hip/hip_bf16.h>
#include <math.h>

typedef __attribute__((ext_vector_type(16))) _Float16 v16h;
typedef __attribute__((ext_vector_type(4)))  _Float16 v4h;
typedef __attribute__((ext_vector_type(8)))  float    v8f;

#define NB   256
#define TT   128
#define II   512
#define HH   512
#define GG   2048                 /* 4*H */
#define NTR  (NB*TT)              /* 32768 rows of (n,t) */
#define NTH  ((size_t)NTR*HH)     /* 16,777,216 */

static __device__ __forceinline__ v8f zero8() {
    v8f z = {0.f,0.f,0.f,0.f,0.f,0.f,0.f,0.f};
    return z;
}
static __device__ __forceinline__ v8f wmma_f16(v16h a, v16h b, v8f c) {
    return __builtin_amdgcn_wmma_f32_16x16x32_f16(false, a, false, b, (short)0, c, false, false);
}
static __device__ __forceinline__ float sigmoidf_(float x) {
    return 1.0f / (1.0f + expf(-x));
}
static __device__ __forceinline__ float mishf_(float x) {
    float sp = (x > 20.0f) ? x : log1pf(expf(x));
    return x * tanhf(sp);
}
// Forward map: K index inside a 32-wide K-tile for the A (16x32 f16) layout.
// lane = m + 16*half; halves j=0..15 -> VGPR v=j/2, koff=j&1.
static __device__ __forceinline__ int a_k_of(int lane, int j) {
    int v = j >> 1, koff = j & 1, half = lane >> 4;
    int base = (v < 4) ? (2 * v) : (16 + 2 * (v - 4));
    return base + 8 * half + koff;
}
// Inverse map: K-in-tile kk (0..31) -> (half, j) of the A layout.
static __device__ __forceinline__ void a_inv_of(int kk, int* half, int* j) {
    int h  = (kk >> 3) & 1;          // K in {8..15}|{24..31} lives in lanes 16..31
    int ku = kk & 0x17;              // drop the half bit
    int v  = (ku < 8) ? (ku >> 1) : (4 + ((ku & 7) >> 1));
    *half = h;
    *j    = 2 * v + (ku & 1);
}

// ---------------- Phase 0: pack weights (f32 row-major -> f16 WMMA-B layout) --
// B tile (32x16): lane = n + 16*khalf (n = col), 16 halves hold K = (j&7) + 8*khalf + 16*(j>>3).
// dst element index = ((nt*ktiles + kt)*32 + lane)*16 + j
__global__ void pack_b_kernel(const float* __restrict__ W, _Float16* __restrict__ dst,
                              int ntiles, int ktiles, int K) {
    int total = ntiles * ktiles * 512;
    for (int idx = blockIdx.x * blockDim.x + threadIdx.x; idx < total;
         idx += gridDim.x * blockDim.x) {
        int j    = idx & 15;
        int lane = (idx >> 4) & 31;
        int tile = idx >> 9;
        int kt   = tile % ktiles;
        int nt   = tile / ktiles;
        int n    = nt * 16 + (lane & 15);
        int k    = kt * 32 + ((j >> 3) << 4) + ((lane >> 4) << 3) + (j & 7);
        dst[idx] = (_Float16)W[n * K + k];
    }
}

__global__ void bias_kernel(const float* __restrict__ bi, const float* __restrict__ bh,
                            float* __restrict__ b) {
    int i = blockIdx.x * blockDim.x + threadIdx.x;
    if (i < GG) b[i] = bi[i] + bh[i];
}

// ---------------- Phase 1: gate preactivations, gate-interleaved f16 ---------
// Gx[row][hcol*4 + q] = (x @ W_ih^T + b_ih + b_hh)[row][q*512 + hcol]
// Block: 512 threads (16 waves), 64 rows x 512 gate-cols.
// Wave w: mtile = w&3, col-quarter ch = w>>2 (128 cols = 8 tiles, 64 acc VGPRs).
__global__ void xgemm_kernel(const float* __restrict__ x, const _Float16* __restrict__ WihP,
                             const float* __restrict__ bias, _Float16* __restrict__ Gx) {
    __shared__ alignas(32) _Float16 As[4 * 16 * 32 * 16]; // [mt][kt][lane][j], 64KB
    const int tid = threadIdx.x;
    const int mb  = blockIdx.x;  // 64-row block
    const int cg  = blockIdx.y;  // 512-gate-col group (0..3)
    for (int idx = tid; idx < 4 * 8192; idx += 512) {
        int j = idx & 15, ln = (idx >> 4) & 31, kt = (idx >> 9) & 15, mt = idx >> 13;
        int m   = ln & 15;
        int k   = kt * 32 + a_k_of(ln, j);
        int row = mb * 64 + mt * 16 + m;
        As[idx] = (_Float16)x[(size_t)row * II + k];
    }
    __syncthreads();
    const int w = tid >> 5, lane = tid & 31;
    const int mt = w & 3, ch = w >> 2;
    v8f acc[8];
#pragma unroll
    for (int i = 0; i < 8; ++i) acc[i] = zero8();
    for (int kt = 0; kt < 16; ++kt) {
        v16h a = *(const v16h*)&As[((mt * 16 + kt) * 32 + lane) * 16];
#pragma unroll
        for (int cb = 0; cb < 8; ++cb) {
            int nt = cg * 32 + ch * 8 + cb;
            v16h b = ((const v16h*)WihP)[(nt * 16 + kt) * 32 + lane];
            acc[cb] = wmma_f16(a, b, acc[cb]);
        }
    }
    const int n0 = lane & 15, half = lane >> 4;
#pragma unroll
    for (int cb = 0; cb < 8; ++cb) {
        int gcol = cg * 512 + ch * 128 + cb * 16 + n0;   // 0..2047
        int q    = gcol >> 9;                            // gate index
        int hcol = gcol & 511;
        float bv = bias[gcol];
#pragma unroll
        for (int r = 0; r < 8; ++r) {
            int row = mb * 64 + mt * 16 + r + 8 * half;
            Gx[(size_t)row * GG + hcol * 4 + q] = (_Float16)(acc[cb][r] + bv);
        }
    }
}

// ---------------- Phase 2: persistent recurrence. 16 WGs x 16 batch rows. ----
// 512 threads = 16 waves; wave w owns h-columns [w*32, w*32+32) (2 tiles/gate
// pair -> 8 acc tiles). c lives in accumulator-layout registers. h state lives
// only as the f16 WMMA-A operand (ping-pong LDS buffers); the update stage
// writes h_new * reset(t+1) straight into the next A buffer.
__global__ void lstm_rec_kernel(const int* __restrict__ is_init,
                                const float* __restrict__ hx, const float* __restrict__ cx,
                                const _Float16* __restrict__ WhhP,
                                const _Float16* __restrict__ Gx,
                                _Float16* __restrict__ hs,       // f16, out slot0
                                float* __restrict__ hT, float* __restrict__ cT) {
    __shared__ alignas(32) _Float16 hA[2][16 * 32 * 16]; // 2 x 16KB ping-pong
    __shared__ float rstC[16];
    __shared__ float rstN[16];
    const int tid = threadIdx.x;
    const int w = tid >> 5, lane = tid & 31;
    const int n0 = lane & 15, half = lane >> 4;
    const int rowbase = blockIdx.x * 16;

    // initial pack: hA[0] = hx[:,0,:] * (1 - is_init[:,0])
    for (int idx = tid; idx < 16 * 32 * 16; idx += 512) {
        int j = idx & 15, ln = (idx >> 4) & 31, kt = idx >> 9;
        int m = ln & 15;
        int k = kt * 32 + a_k_of(ln, j);
        float r0 = 1.0f - (float)is_init[(rowbase + m) * TT + 0];
        hA[0][idx] = (_Float16)(hx[(size_t)(rowbase + m) * TT * HH + k] * r0);
    }
    v8f creg[2];
#pragma unroll
    for (int cb = 0; cb < 2; ++cb)
#pragma unroll
        for (int r = 0; r < 8; ++r) {
            int m = r + 8 * half, col = w * 32 + cb * 16 + n0;
            creg[cb][r] = cx[(size_t)(rowbase + m) * TT * HH + col]; // cx[:,0,:]
        }

    for (int t = 0; t < TT; ++t) {
        __syncthreads();   // prev update (hA-next writes, rst reads) fully done
        if (tid < 16) {
            rstC[tid] = 1.0f - (float)is_init[(rowbase + tid) * TT + t];
            rstN[tid] = (t + 1 < TT)
                        ? 1.0f - (float)is_init[(rowbase + tid) * TT + t + 1] : 1.0f;
        }
        __syncthreads();   // hA[t&1] + rst arrays ready
        const _Float16* Ab = hA[t & 1];
        _Float16*       An = hA[(t + 1) & 1];
        v8f acc[8];
#pragma unroll
        for (int i = 0; i < 8; ++i) acc[i] = zero8();
        for (int kt = 0; kt < 16; ++kt) {
            v16h a = *(const v16h*)&Ab[(kt * 32 + lane) * 16];
#pragma unroll
            for (int i = 0; i < 8; ++i) {
                int nt = (i >> 1) * 32 + w * 2 + (i & 1);   // q*32 + w*2 + cb
                v16h b = ((const v16h*)WhhP)[(nt * 16 + kt) * 32 + lane];
                acc[i] = wmma_f16(a, b, acc[i]);
            }
        }
#pragma unroll
        for (int cb = 0; cb < 2; ++cb) {
            int col = w * 32 + cb * 16 + n0;
            int kt2 = col >> 5;
            int halfp, j2;
            a_inv_of(col & 31, &halfp, &j2);
#pragma unroll
            for (int r = 0; r < 8; ++r) {
                int m = r + 8 * half;
                size_t rowoff = (size_t)(rowbase + m) * TT + t;
                v4h g4 = ((const v4h*)Gx)[rowoff * HH + col];  // i,f,g,o contiguous
                float gi = acc[0 * 2 + cb][r] + (float)g4[0];
                float gf = acc[1 * 2 + cb][r] + (float)g4[1];
                float gg = acc[2 * 2 + cb][r] + (float)g4[2];
                float go = acc[3 * 2 + cb][r] + (float)g4[3];
                float cold = creg[cb][r] * rstC[m];
                float cn = sigmoidf_(gf) * cold + sigmoidf_(gi) * tanhf(gg);
                float hn = sigmoidf_(go) * tanhf(cn);
                creg[cb][r] = cn;
                An[(kt2 * 32 + (m + 16 * halfp)) * 16 + j2] = (_Float16)(hn * rstN[m]);
                hs[rowoff * HH + col] = (_Float16)hn;
                if (t == TT - 1) {
                    hT[(rowbase + m) * HH + col] = hn;
                    cT[(rowbase + m) * HH + col] = cn;
                }
            }
        }
    }
}

// ---------------- Phase 3: out = mish(hs @ W_out^T + b_out) -> slot1 ---------
// 512 threads, 64 rows x 512 cols; wave w: mt = w&3, ch = w>>2 (8 tiles).
__global__ void proj_kernel(const _Float16* __restrict__ hsv, const _Float16* __restrict__ WoutP,
                            const float* __restrict__ b_out, float* __restrict__ outTmp) {
    __shared__ alignas(32) _Float16 As[4 * 16 * 32 * 16];
    const int tid = threadIdx.x;
    const int mb  = blockIdx.x;
    for (int idx = tid; idx < 4 * 8192; idx += 512) {
        int j = idx & 15, ln = (idx >> 4) & 31, kt = (idx >> 9) & 15, mt = idx >> 13;
        int m   = ln & 15;
        int k   = kt * 32 + a_k_of(ln, j);
        int row = mb * 64 + mt * 16 + m;
        As[idx] = hsv[(size_t)row * HH + k];           // already f16
    }
    __syncthreads();
    const int w = tid >> 5, lane = tid & 31;
    const int mt = w & 3, ch = w >> 2;
    v8f acc[8];
#pragma unroll
    for (int i = 0; i < 8; ++i) acc[i] = zero8();
    for (int kt = 0; kt < 16; ++kt) {
        v16h a = *(const v16h*)&As[((mt * 16 + kt) * 32 + lane) * 16];
#pragma unroll
        for (int cb = 0; cb < 8; ++cb) {
            int nt = ch * 8 + cb;
            v16h b = ((const v16h*)WoutP)[(nt * 16 + kt) * 32 + lane];
            acc[cb] = wmma_f16(a, b, acc[cb]);
        }
    }
    const int n0 = lane & 15, half = lane >> 4;
#pragma unroll
    for (int cb = 0; cb < 8; ++cb) {
        int col  = ch * 128 + cb * 16 + n0;
        float bv = b_out[col];
#pragma unroll
        for (int r = 0; r < 8; ++r) {
            int row = mb * 64 + mt * 16 + r + 8 * half;
            outTmp[(size_t)row * HH + col] = mishf_(acc[cb][r] + bv);
        }
    }
}

// ---------------- Phase 4: move result to slot0, broadcast hT/cT ------------
__global__ void finalize_kernel(float* out, const float* __restrict__ hT,
                                const float* __restrict__ cT) {
    size_t i = (size_t)blockIdx.x * blockDim.x + threadIdx.x;
    if (i >= NTH) return;
    int n = (int)(i / ((size_t)TT * HH));
    int k = (int)(i % HH);
    float r = out[NTH + i];                    // projection result (slot1)
    out[i] = r;                                // output  -> slot0
    out[NTH + i]     = hT[n * HH + k];         // h_rep   -> slot1
    out[2 * NTH + i] = cT[n * HH + k];         // c_rep   -> slot2
}

extern "C" void kernel_launch(void* const* d_in, const int* in_sizes, int n_in,
                              void* d_out, int out_size, void* d_ws, size_t ws_size,
                              hipStream_t stream) {
    const float* x    = (const float*)d_in[0];
    const int*   isin = (const int*)d_in[1];
    const float* hx   = (const float*)d_in[2];
    const float* cx   = (const float*)d_in[3];
    const float* Wih  = (const float*)d_in[4];
    const float* Whh  = (const float*)d_in[5];
    const float* bih  = (const float*)d_in[6];
    const float* bhh  = (const float*)d_in[7];
    const float* Wout = (const float*)d_in[8];
    const float* bout = (const float*)d_in[9];
    float* out = (float*)d_out;
    char*  ws  = (char*)d_ws;

    _Float16* WihP  = (_Float16*)(ws);                 // 2 MB
    _Float16* WhhP  = (_Float16*)(ws + 2097152);       // 2 MB
    _Float16* WoutP = (_Float16*)(ws + 4194304);       // 0.5 MB
    float*    bias  = (float*)(ws + 4718592);          // 8 KB
    float*    hT    = (float*)(ws + 4726784);          // 0.5 MB
    float*    cT    = (float*)(ws + 5251072);          // 0.5 MB
    _Float16* Gx    = (_Float16*)(out + NTH);          // 128 MB, lives in slots 1+2
    _Float16* hsF   = (_Float16*)out;                  // 32 MB, lives in slot 0
    float*    slot1 = out + NTH;

    pack_b_kernel<<<1024, 256, 0, stream>>>(Wih,  WihP, 128, 16, 512);
    pack_b_kernel<<<1024, 256, 0, stream>>>(Whh,  WhhP, 128, 16, 512);
    pack_b_kernel<<<512,  256, 0, stream>>>(Wout, WoutP, 32, 16, 512);
    bias_kernel<<<8, 256, 0, stream>>>(bih, bhh, bias);

    xgemm_kernel<<<dim3(NTR / 64, 4), 512, 0, stream>>>(x, WihP, bias, Gx);
    lstm_rec_kernel<<<NB / 16, 512, 0, stream>>>(isin, hx, cx, WhhP, Gx, hsF, hT, cT);
    proj_kernel<<<NTR / 64, 512, 0, stream>>>(hsF, WoutP, bout, slot1);
    finalize_kernel<<<(unsigned)(NTH / 256), 256, 0, stream>>>(out, hT, cT);
}
